// MaskedAttentionEmb_72919954751582
// MI455X (gfx1250) — matrix operand order repaired and checked by reading, not compiled, not run on hardware
//
#include <hip/hip_runtime.h>
#include <hip/hip_bf16.h>
#include <math.h>

#define B_      32
#define S_      128
#define EMB_    128
#define HID_    768
#define FF_     3072
#define LAYERS_ 12
#define NH_     12
#define HD_     64
#define CW_     3
#define M_      (B_*S_)        // 4096 rows of activations

typedef __attribute__((ext_vector_type(16))) _Float16 v16h;
typedef __attribute__((ext_vector_type(8)))  _Float16 v8h;
typedef __attribute__((ext_vector_type(8)))  float    v8f;
typedef __attribute__((ext_vector_type(4)))  int      v4i;

#define GLOBAL_AS __attribute__((address_space(1)))
#define LDS_AS    __attribute__((address_space(3)))

// Async global->LDS copy support (CDNA5). Guarded so both toolchains compile.
#if defined(__has_builtin)
# if __has_builtin(__builtin_amdgcn_global_load_async_to_lds_b128)
#  define HAVE_ASYNC_LDS 1
# endif
#endif
#ifndef HAVE_ASYNC_LDS
# define HAVE_ASYNC_LDS 0
#endif

__device__ __forceinline__ void wait_async_zero() {
#if defined(__has_builtin) && __has_builtin(__builtin_amdgcn_s_wait_asynccnt)
    __builtin_amdgcn_s_wait_asynccnt(0);
#else
    asm volatile("s_wait_asynccnt 0x0" ::: "memory");
#endif
}

__device__ __forceinline__ float gelu_exact(float x) {
    return 0.5f * x * (1.0f + erff(x * 0.70710678118654752440f));
}

// ---------------------------------------------------------------------------
// f32 -> f16 cast (weights only; grid-stride)
// ---------------------------------------------------------------------------
__global__ void cast_f32_to_f16(const float* __restrict__ src,
                                _Float16* __restrict__ dst, size_t n) {
    size_t i = (size_t)blockIdx.x * blockDim.x + threadIdx.x;
    size_t stride = (size_t)gridDim.x * blockDim.x;
    for (; i < n; i += stride) dst[i] = (_Float16)src[i];
}

// Gather embedding rows by token id and cast to f16: A[m,k] = embed[ids[m],k]
__global__ void gather_cast_embed(const int* __restrict__ ids,
                                  const float* __restrict__ embed,
                                  _Float16* __restrict__ dst) {
    int m = blockIdx.x;          // 0..M_-1
    int k = threadIdx.x;         // 0..EMB_-1 (blockDim = 128)
    dst[(size_t)m * EMB_ + k] = (_Float16)embed[(size_t)ids[m] * EMB_ + k];
}

// Sinusoidal positional encoding pe[s, d]
__global__ void pe_kernel(float* __restrict__ pe) {
    int i = blockIdx.x * blockDim.x + threadIdx.x;   // S_*HID_ threads
    int s = i / HID_;
    int d = i % HID_;
    int j2 = (d >> 1) << 1;                          // 2*j
    float div = expf((float)j2 * (-9.210340371976184f / (float)HID_)); // ln(1e4)
    float arg = (float)s * div;
    pe[i] = (d & 1) ? cosf(arg) : sinf(arg);
}

// ---------------------------------------------------------------------------
// WMMA GEMM:  out = A[M,K] @ W[N,K]^T + bias (+residual) (gelu) (+pe)
// A, W are f16 (K contiguous for both). Optional f32 and/or f16 outputs.
// Block = 256 threads = 8 waves. Wave tile 32x64 (8 wmma / K-step of 32).
// Block tile = 128(M) x 128(N).  grid = (N/128, M/128).
// flags bit0: exact GELU after bias(+residual).
// Async path: double-buffered LDS staging of the 128x32 A/B tiles via
// GLOBAL_LOAD_ASYNC_TO_LDS_B128 (ASYNCcnt), DMA of step k+1 overlaps WMMA of k.
// ---------------------------------------------------------------------------
__global__ __launch_bounds__(256) void gemm_wmma_f16(
    const _Float16* __restrict__ A, const _Float16* __restrict__ W,
    const float* __restrict__ bias, const float* __restrict__ residual,
    const float* __restrict__ pe, float* __restrict__ C,
    _Float16* __restrict__ C16, int M, int N, int K, int flags)
{
    const int lane = threadIdx.x & 31;
    const int wid  = threadIdx.x >> 5;
    const int r    = lane & 15;     // row within 16 (A: M, B: N, C: N)
    const int hs   = lane >> 4;     // lane-half select

    const int tileM = blockIdx.y * 128 + (wid >> 1) * 32;
    const int tileN = blockIdx.x * 128 + (wid & 1) * 64;

    v8f c[2][4] = {};

#if HAVE_ASYNC_LDS
    // ---- LDS-staged, double-buffered path (CDNA5 async copies) ----
    constexpr int LDP = 40;                       // padded row (halfs): bank-quad free
    __shared__ _Float16 lA[2][128 * LDP];
    __shared__ _Float16 lB[2][128 * LDP];
    const int tid = threadIdx.x;
    const int tileMBase = blockIdx.y * 128;
    const int tileNBase = blockIdx.x * 128;

    auto issue = [&](int k, int buf) {
#pragma unroll
        for (int j = 0; j < 2; ++j) {
            int cch = tid + 256 * j;              // 0..511 chunks of 16B
            int row = cch >> 2, seg = cch & 3;    // 128 rows x 4 segs
            const _Float16* gA = A + (size_t)(tileMBase + row) * K + k + seg * 8;
            const _Float16* gB = W + (size_t)(tileNBase + row) * K + k + seg * 8;
            __builtin_amdgcn_global_load_async_to_lds_b128(
                (GLOBAL_AS v4i*)gA,
                (LDS_AS v4i*)&lA[buf][row * LDP + seg * 8],
                0, 0);
            __builtin_amdgcn_global_load_async_to_lds_b128(
                (GLOBAL_AS v4i*)gB,
                (LDS_AS v4i*)&lB[buf][row * LDP + seg * 8],
                0, 0);
        }
    };

    issue(0, 0);
    const int nk = K >> 5;
    const int rA0 = (wid >> 1) * 32;
    const int rB0 = (wid & 1) * 64;
    for (int ks = 0; ks < nk; ++ks) {
        const int buf = ks & 1;
        wait_async_zero();                 // my copies for `buf` are done
        __syncthreads();                   // everyone's copies done; prior reads done
        if (ks + 1 < nk) issue((ks + 1) * 32, buf ^ 1);

        v16h a[2], b[4];
#pragma unroll
        for (int mi = 0; mi < 2; ++mi) {
            const v8h* p = (const v8h*)&lA[buf][(rA0 + 16 * mi + r) * LDP + hs * 8];
            a[mi] = __builtin_shufflevector(p[0], p[2], 0,1,2,3,4,5,6,7,8,9,10,11,12,13,14,15);
        }
#pragma unroll
        for (int ni = 0; ni < 4; ++ni) {
            const v8h* p = (const v8h*)&lB[buf][(rB0 + 16 * ni + r) * LDP + hs * 8];
            b[ni] = __builtin_shufflevector(p[0], p[2], 0,1,2,3,4,5,6,7,8,9,10,11,12,13,14,15);
        }
#pragma unroll
        for (int mi = 0; mi < 2; ++mi)
#pragma unroll
            for (int ni = 0; ni < 4; ++ni)
                c[mi][ni] = __builtin_amdgcn_wmma_f32_16x16x32_f16(
                    false, a[mi], false, b[ni], (short)0, c[mi][ni], false, false);
    }
#else
    // ---- direct-load path (known-good fallback) ----
    for (int k = 0; k < K; k += 32) {
        v16h a[2];
#pragma unroll
        for (int mi = 0; mi < 2; ++mi) {
            const v8h* p = (const v8h*)(A + (size_t)(tileM + 16 * mi + r) * K + k + hs * 8);
            v8h lo = p[0];
            v8h hi = p[2];   // +16 halfs
            a[mi] = __builtin_shufflevector(lo, hi, 0,1,2,3,4,5,6,7,8,9,10,11,12,13,14,15);
        }
        v16h b[4];
#pragma unroll
        for (int ni = 0; ni < 4; ++ni) {
            const v8h* p = (const v8h*)(W + (size_t)(tileN + 16 * ni + r) * K + k + hs * 8);
            v8h lo = p[0];
            v8h hi = p[2];
            b[ni] = __builtin_shufflevector(lo, hi, 0,1,2,3,4,5,6,7,8,9,10,11,12,13,14,15);
        }
#pragma unroll
        for (int mi = 0; mi < 2; ++mi)
#pragma unroll
            for (int ni = 0; ni < 4; ++ni)
                c[mi][ni] = __builtin_amdgcn_wmma_f32_16x16x32_f16(
                    false, a[mi], false, b[ni], (short)0, c[mi][ni], false, false);
    }
#endif

    // Epilogue. C layout: VGPR v, lanes 0-15 -> m=v, lanes 16-31 -> m=v+8; n=lane&15.
#pragma unroll
    for (int mi = 0; mi < 2; ++mi) {
#pragma unroll
        for (int ni = 0; ni < 4; ++ni) {
            const int col = tileN + 16 * ni + r;
            const float bv = bias[col];
#pragma unroll
            for (int v = 0; v < 8; ++v) {
                const int row = tileM + 16 * mi + v + 8 * hs;
                float val = c[mi][ni][v] + bv;
                if (residual) val += residual[(size_t)row * N + col];
                if (flags & 1) val = gelu_exact(val);
                if (pe)       val += pe[(size_t)(row & (S_ - 1)) * N + col];
                if (C)   C[(size_t)row * N + col] = val;
                if (C16) C16[(size_t)row * N + col] = (_Float16)val;
            }
        }
    }
}

// ---------------------------------------------------------------------------
// Banded attention: window |i-j| <= CW_, NH_ heads of HD_=64.
// One wave per (b, h, i). Lane covers dims d and d+32. Output written as f16
// (it is only ever consumed as the A operand of the next WMMA GEMM).
// ---------------------------------------------------------------------------
__global__ __launch_bounds__(256) void band_attn(const float* __restrict__ qkv,
                                                 _Float16* __restrict__ out) {
    const int lane = threadIdx.x & 31;
    const int wid  = threadIdx.x >> 5;
    const int flat = blockIdx.x * 8 + wid;       // (b*NH + h)*S + i
    const int i  = flat & (S_ - 1);
    const int bh = flat >> 7;
    const int b  = bh / NH_;
    const int h  = bh % NH_;
    const size_t rs = 3 * HID_;

    const float* qrow = qkv + (size_t)(b * S_ + i) * rs + h * HD_;
    const float q0 = qrow[lane], q1 = qrow[lane + 32];

    float sc[2 * CW_ + 1];
#pragma unroll
    for (int t = 0; t < 2 * CW_ + 1; ++t) {
        int j = i - CW_ + t;
        float p = -__builtin_inff();
        if (j >= 0 && j < S_) {
            const float* krow = qkv + (size_t)(b * S_ + j) * rs + HID_ + h * HD_;
            p = q0 * krow[lane] + q1 * krow[lane + 32];
#pragma unroll
            for (int off = 16; off >= 1; off >>= 1) p += __shfl_xor(p, off, 32);
            p *= 0.125f;   // 1/sqrt(64)
        }
        sc[t] = p;
    }

    float mx = sc[0];
#pragma unroll
    for (int t = 1; t < 2 * CW_ + 1; ++t) mx = fmaxf(mx, sc[t]);
    float w[2 * CW_ + 1], den = 0.f;
#pragma unroll
    for (int t = 0; t < 2 * CW_ + 1; ++t) {
        w[t] = (sc[t] == -__builtin_inff()) ? 0.f : expf(sc[t] - mx);
        den += w[t];
    }
    const float inv = 1.f / den;

    float o0 = 0.f, o1 = 0.f;
#pragma unroll
    for (int t = 0; t < 2 * CW_ + 1; ++t) {
        if (w[t] != 0.f) {
            int j = i - CW_ + t;
            const float* vrow = qkv + (size_t)(b * S_ + j) * rs + 2 * HID_ + h * HD_;
            float aw = w[t] * inv;
            o0 += aw * vrow[lane];
            o1 += aw * vrow[lane + 32];
        }
    }
    _Float16* orow = out + (size_t)(b * S_ + i) * HID_ + h * HD_;
    orow[lane]      = (_Float16)o0;
    orow[lane + 32] = (_Float16)o1;
}

// ---------------------------------------------------------------------------
// Row LayerNorm. Dual output: f32 (future residual) + f16 (next GEMM operand).
// ---------------------------------------------------------------------------
__global__ __launch_bounds__(256) void ln_kernel(const float* __restrict__ x,
                                                 const float* __restrict__ g,
                                                 const float* __restrict__ b,
                                                 float* __restrict__ out,
                                                 _Float16* __restrict__ out16, int N) {
    __shared__ float s1[256], s2[256];
    const int row = blockIdx.x;
    const int t = threadIdx.x;
    float sum = 0.f, sq = 0.f;
    for (int c = t; c < N; c += 256) {
        float v = x[(size_t)row * N + c];
        sum += v; sq += v * v;
    }
    s1[t] = sum; s2[t] = sq;
    __syncthreads();
    for (int off = 128; off > 0; off >>= 1) {
        if (t < off) { s1[t] += s1[t + off]; s2[t] += s2[t + off]; }
        __syncthreads();
    }
    const float mean = s1[0] / (float)N;
    const float var  = s2[0] / (float)N - mean * mean;
    const float inv  = rsqrtf(var + 1e-5f);
    for (int c = t; c < N; c += 256) {
        float v = x[(size_t)row * N + c];
        float y = (v - mean) * inv * g[c] + b[c];
        out[(size_t)row * N + c]   = y;
        out16[(size_t)row * N + c] = (_Float16)y;
    }
}

// L2-normalize rows of a [rows, EMB_] matrix. One wave per row.
__global__ __launch_bounds__(256) void l2norm_rows(const float* __restrict__ src,
                                                   float* __restrict__ dst, int rows) {
    const int lane = threadIdx.x & 31;
    const int row = blockIdx.x * 8 + (threadIdx.x >> 5);
    if (row >= rows) return;
    const float* p = src + (size_t)row * EMB_;
    float e0 = p[lane], e1 = p[lane + 32], e2 = p[lane + 64], e3 = p[lane + 96];
    float s = e0 * e0 + e1 * e1 + e2 * e2 + e3 * e3;
#pragma unroll
    for (int off = 16; off >= 1; off >>= 1) s += __shfl_xor(s, off, 32);
    const float inv = 1.f / sqrtf(s);
    float* q = dst + (size_t)row * EMB_;
    q[lane] = e0 * inv; q[lane + 32] = e1 * inv;
    q[lane + 64] = e2 * inv; q[lane + 96] = e3 * inv;
}

// ---------------------------------------------------------------------------
// Losses
// ---------------------------------------------------------------------------
__global__ void zero_acc(float* acc) { if (threadIdx.x < 4) acc[threadIdx.x] = 0.f; }

__global__ __launch_bounds__(256) void n2v_loss(const float* __restrict__ preds,
                                                const float* __restrict__ negp,
                                                float* __restrict__ acc) {
    const int lane = threadIdx.x & 31;
    const int flat = blockIdx.x * 8 + (threadIdx.x >> 5);
    const int T = S_ - 2 * CW_;               // 122
    if (flat >= B_ * T) return;
    const int b = flat / T, t = flat % T;
    const float* cen = preds + (size_t)(b * S_ + t + CW_) * EMB_;
    const float c0 = cen[lane], c1 = cen[lane + 32],
                c2 = cen[lane + 64], c3 = cen[lane + 96];
    const int offs[6] = {0, 1, 2, CW_ + 1, CW_ + 2, CW_ + 3};
    float spos = 0.f, sneg = 0.f;
#pragma unroll
    for (int oi = 0; oi < 6; ++oi) {
        const int crow = b * S_ + t + offs[oi];
        const float* cp = preds + (size_t)crow * EMB_;
        const float* np = negp  + (size_t)crow * EMB_;
        float dp = c0 * cp[lane] + c1 * cp[lane + 32] + c2 * cp[lane + 64] + c3 * cp[lane + 96];
        float dn = c0 * np[lane] + c1 * np[lane + 32] + c2 * np[lane + 64] + c3 * np[lane + 96];
#pragma unroll
        for (int off = 16; off >= 1; off >>= 1) {
            dp += __shfl_xor(dp, off, 32);
            dn += __shfl_xor(dn, off, 32);
        }
        const float sp = 1.f / (1.f + expf(-dp));
        const float sn = 1.f / (1.f + expf(-dn));
        spos += logf(sp + 1e-8f);
        sneg += logf(1.f - sn + 1e-8f);
    }
    if (lane == 0) { atomicAdd(&acc[0], spos); atomicAdd(&acc[1], sneg); }
}

__global__ __launch_bounds__(256) void recon_loss(const int* __restrict__ seq,
                                                  const float* __restrict__ out_embs,
                                                  const float* __restrict__ preds,
                                                  float* __restrict__ acc) {
    const int lane = threadIdx.x & 31;
    const int flat = blockIdx.x * 8 + (threadIdx.x >> 5);
    const int T = S_ - 2 * CW_;
    if (flat >= B_ * T) return;
    const int b = flat / T, t = flat % T;
    const int pos = t + CW_;
    const int id = seq[b * S_ + pos];
    const float* oe = out_embs + (size_t)id * EMB_;
    float e0 = oe[lane], e1 = oe[lane + 32], e2 = oe[lane + 64], e3 = oe[lane + 96];
    float s = e0 * e0 + e1 * e1 + e2 * e2 + e3 * e3;
#pragma unroll
    for (int off = 16; off >= 1; off >>= 1) s += __shfl_xor(s, off, 32);
    const float inv = 1.f / sqrtf(s);
    const float* tg = preds + (size_t)(b * S_ + pos) * EMB_;
    float d0 = e0 * inv - tg[lane],      d1 = e1 * inv - tg[lane + 32];
    float d2 = e2 * inv - tg[lane + 64], d3 = e3 * inv - tg[lane + 96];
    float sq = d0 * d0 + d1 * d1 + d2 * d2 + d3 * d3;
#pragma unroll
    for (int off = 16; off >= 1; off >>= 1) sq += __shfl_xor(sq, off, 32);
    if (lane == 0) atomicAdd(&acc[2], sq);
}

__global__ void finalize_loss(const float* __restrict__ acc, float* __restrict__ out) {
    if (threadIdx.x == 0) {
        const float nPairs = (float)(B_ * (S_ - 2 * CW_) * 2 * CW_);     // 23424
        out[0] = -(acc[0] / nPairs) - (acc[1] / nPairs);
        out[1] = acc[2] / (float)(B_ * (S_ - 2 * CW_) * EMB_);           // 499712
    }
}

// ---------------------------------------------------------------------------
// Host orchestration
// ---------------------------------------------------------------------------
extern "C" void kernel_launch(void* const* d_in, const int* in_sizes, int n_in,
                              void* d_out, int out_size, void* d_ws, size_t ws_size,
                              hipStream_t stream) {
    (void)in_sizes; (void)n_in; (void)out_size; (void)ws_size;

    const int*   seq      = (const int*)d_in[0];
    const int*   rnd      = (const int*)d_in[1];
    const float* embed    = (const float*)d_in[2];
    const float* proj_w   = (const float*)d_in[3];
    const float* proj_b   = (const float*)d_in[4];
    const float* qkv_w    = (const float*)d_in[5];
    const float* qkv_b    = (const float*)d_in[6];
    const float* attn_ow  = (const float*)d_in[7];
    const float* attn_ob  = (const float*)d_in[8];
    const float* ln1_g    = (const float*)d_in[9];
    const float* ln1_b    = (const float*)d_in[10];
    const float* ff1_w    = (const float*)d_in[11];
    const float* ff1_b    = (const float*)d_in[12];
    const float* ff2_w    = (const float*)d_in[13];
    const float* ff2_b    = (const float*)d_in[14];
    const float* ln2_g    = (const float*)d_in[15];
    const float* ln2_b    = (const float*)d_in[16];
    const float* out_w    = (const float*)d_in[17];
    const float* out_b    = (const float*)d_in[18];
    const float* out_embs = (const float*)d_in[19];
    float* out = (float*)d_out;

    // ---- workspace carve-up (256-byte aligned bump allocator) ----
    char* p = (char*)d_ws;
    auto alloc = [&](size_t bytes) -> char* {
        char* q = p;
        p += (bytes + 255) & ~(size_t)255;
        return q;
    };
    const size_t n_qkv_w = (size_t)LAYERS_ * 3 * HID_ * HID_;
    const size_t n_ao_w  = (size_t)LAYERS_ * HID_ * HID_;
    const size_t n_ff1_w = (size_t)LAYERS_ * FF_ * HID_;
    const size_t n_ff2_w = (size_t)LAYERS_ * HID_ * FF_;
    const size_t n_proj  = (size_t)HID_ * EMB_;
    const size_t n_outw  = (size_t)EMB_ * HID_;

    _Float16* hW_qkv  = (_Float16*)alloc(n_qkv_w * 2);
    _Float16* hW_attn = (_Float16*)alloc(n_ao_w  * 2);
    _Float16* hW_ff1  = (_Float16*)alloc(n_ff1_w * 2);
    _Float16* hW_ff2  = (_Float16*)alloc(n_ff2_w * 2);
    _Float16* hW_proj = (_Float16*)alloc(n_proj  * 2);
    _Float16* hW_out  = (_Float16*)alloc(n_outw  * 2);

    _Float16* hX   = (_Float16*)alloc((size_t)M_ * HID_ * 2);   // f16 activations
    _Float16* hAO  = (_Float16*)alloc((size_t)M_ * HID_ * 2);   // f16 attention out
    _Float16* hH   = (_Float16*)alloc((size_t)M_ * FF_  * 2);   // f16 FF hidden
    _Float16* hEmb = (_Float16*)alloc((size_t)M_ * EMB_ * 2);   // f16 gathered embeds

    float* x    = (float*)alloc((size_t)M_ * HID_ * 4);
    float* qkvB = (float*)alloc((size_t)M_ * 3 * HID_ * 4);
    float* tbuf = (float*)alloc((size_t)M_ * HID_ * 4);
    float* prds = (float*)alloc((size_t)M_ * EMB_ * 4);
    float* negp = (float*)alloc((size_t)M_ * EMB_ * 4);
    float* pe   = (float*)alloc((size_t)S_ * HID_ * 4);
    float* acc  = (float*)alloc(4 * 4);

    auto cast = [&](const float* src, _Float16* dst, size_t n) {
        cast_f32_to_f16<<<4096, 256, 0, stream>>>(src, dst, n);
    };
    auto gemm = [&](const _Float16* A, const _Float16* W, const float* bias,
                    const float* res, const float* pe_, float* C, _Float16* C16,
                    int M, int N, int K, int flags) {
        dim3 g(N / 128, M / 128);
        gemm_wmma_f16<<<g, 256, 0, stream>>>(A, W, bias, res, pe_, C, C16, M, N, K, flags);
    };

    // one-time per launch: weights -> f16 (~170MB: fits L2 for encode #2), PE table
    cast(qkv_w,   hW_qkv,  n_qkv_w);
    cast(attn_ow, hW_attn, n_ao_w);
    cast(ff1_w,   hW_ff1,  n_ff1_w);
    cast(ff2_w,   hW_ff2,  n_ff2_w);
    cast(proj_w,  hW_proj, n_proj);
    cast(out_w,   hW_out,  n_outw);
    pe_kernel<<<(S_ * HID_) / 256, 256, 0, stream>>>(pe);

    auto encode = [&](const int* ids, float* preds_out) {
        gather_cast_embed<<<M_, EMB_, 0, stream>>>(ids, embed, hEmb);
        // x = gelu(embed[ids] @ proj_w^T + proj_b) + pe   (f32 + f16 outputs)
        gemm(hEmb, hW_proj, proj_b, nullptr, pe, x, hX, M_, HID_, EMB_, 1);
        for (int li = 0; li < LAYERS_; ++li) {
            gemm(hX, hW_qkv + (size_t)li * 3 * HID_ * HID_, qkv_b + li * 3 * HID_,
                 nullptr, nullptr, qkvB, nullptr, M_, 3 * HID_, HID_, 0);
            band_attn<<<B_ * NH_ * S_ / 8, 256, 0, stream>>>(qkvB, hAO);
            gemm(hAO, hW_attn + (size_t)li * HID_ * HID_, attn_ob + li * HID_,
                 x, nullptr, tbuf, nullptr, M_, HID_, HID_, 0);       // t = x + o@W^T + b
            ln_kernel<<<M_, 256, 0, stream>>>(tbuf, ln1_g + li * HID_, ln1_b + li * HID_,
                                              x, hX, HID_);
            gemm(hX, hW_ff1 + (size_t)li * FF_ * HID_, ff1_b + li * FF_,
                 nullptr, nullptr, nullptr, hH, M_, FF_, HID_, 1);    // f16-only output
            gemm(hH, hW_ff2 + (size_t)li * HID_ * FF_, ff2_b + li * HID_,
                 x, nullptr, tbuf, nullptr, M_, HID_, FF_, 0);        // t = x + h@W2^T + b
            ln_kernel<<<M_, 256, 0, stream>>>(tbuf, ln2_g + li * HID_, ln2_b + li * HID_,
                                              x, hX, HID_);
        }
        gemm(hX, hW_out, out_b, nullptr, nullptr, tbuf, nullptr, M_, EMB_, HID_, 0);
        l2norm_rows<<<M_ / 8, 256, 0, stream>>>(tbuf, preds_out, M_);
    };

    encode(seq, prds);
    encode(rnd, negp);

    zero_acc<<<1, 32, 0, stream>>>(acc);
    const int nBT = B_ * (S_ - 2 * CW_);
    n2v_loss<<<(nBT + 7) / 8, 256, 0, stream>>>(prds, negp, acc);
    recon_loss<<<(nBT + 7) / 8, 256, 0, stream>>>(seq, out_embs, prds, acc);
    finalize_loss<<<1, 32, 0, stream>>>(acc, out);
}
